// Head_85864986181949
// MI455X (gfx1250) — matrix operand order repaired and checked by reading, not compiled, not run on hardware
//
#include <hip/hip_runtime.h>
#include <hip/hip_bf16.h>
#include <cstdint>

typedef __attribute__((ext_vector_type(16))) _Float16 v16h;
typedef __attribute__((ext_vector_type(8)))  float    v8f;

#define HEAD  64
#define EMB   1024
#define BATCH 8
#define SEQ   2048

// ---- CDNA5 async global->LDS DMA helpers (ASYNCcnt-tracked) -----------------
__device__ __forceinline__ uint32_t lds_lo32(const void* p) {
    // flat->LDS hardware mapping truncates to addr[31:0]
    return (uint32_t)(uintptr_t)p;
}
__device__ __forceinline__ void async_b128(uint32_t lds_off, uint64_t gaddr) {
    asm volatile("global_load_async_to_lds_b128 %0, %1, off"
                 :: "v"(lds_off), "v"(gaddr) : "memory");
}
__device__ __forceinline__ void wait_async0() {
    asm volatile("s_wait_asynccnt 0x0" ::: "memory");
}

// ---- Fragment builders (CDNA5 wave32 WMMA 16x16x32 f16 layouts) -------------
// A 16x32: lanes 0-15 -> M=lane, K {0..7,16..23}; lanes 16-31 -> K {8..15,24..31}.
__device__ __forceinline__ v16h frag_a(const _Float16* t, int stride) {
    const int l  = threadIdx.x & 31;
    const int m  = l & 15;
    const int kb = (l >> 4) << 3;
    v16h a;
#pragma unroll
    for (int r = 0; r < 8; ++r) {
        int k = ((r & 4) << 2) + ((r & 3) << 1) + kb;
        a[2*r]   = t[m*stride + k];
        a[2*r+1] = t[m*stride + k + 1];
    }
    return a;
}
// B 32x16: lanes 0-15 -> N=lane, K {0..7,16..23}; lanes 16-31 -> K {8..15,24..31}.
// tile stored N-major: (k,n) at t[n*nstride + k].
__device__ __forceinline__ v16h frag_b(const _Float16* t, int nstride) {
    const int l  = threadIdx.x & 31;
    const int n  = l & 15;
    const int kb = (l >> 4) << 3;
    v16h b;
#pragma unroll
    for (int r = 0; r < 8; ++r) {
        int k = ((r & 4) << 2) + ((r & 3) << 1) + kb;
        b[2*r]   = t[n*nstride + k];
        b[2*r+1] = t[n*nstride + k + 1];
    }
    return b;
}
// C/D f32 16x16: VGPR r holds row M = r + 8*(lane>=16), col N = lane&15.

// ---- Kernel 1: fused QKV projection, 4 waves/block share the W staging ------
__global__ void __launch_bounds__(128)
qkv_proj_kernel(const float* __restrict__ x,
                const float* __restrict__ Wq,
                const float* __restrict__ Wk,
                const float* __restrict__ Wv,
                _Float16* __restrict__ qh,
                _Float16* __restrict__ kh,
                _Float16* __restrict__ vh)
{
    __shared__ alignas(16) _Float16 xs[4][16 * 32];   // per-wave x tile [m][k]
    __shared__ alignas(16) _Float16 wls[3][64 * 32];  // shared W chunks, N-major

    const int tid  = threadIdx.x;
    const int wid  = tid >> 5;
    const int lane = tid & 31;
    const int b    = blockIdx.x / (SEQ / 64);
    const int t0   = (blockIdx.x % (SEQ / 64)) * 64;
    const int tw   = t0 + wid * 16;                   // this wave's 16 rows
    const float* wg[3] = {Wq, Wk, Wv};

    const v8f zero = {0.f,0.f,0.f,0.f,0.f,0.f,0.f,0.f};
    v8f acc[3][4];
#pragma unroll
    for (int w = 0; w < 3; ++w)
#pragma unroll
        for (int nt = 0; nt < 4; ++nt) acc[w][nt] = zero;

    for (int e0 = 0; e0 < EMB; e0 += 32) {
        // per-wave x tile 16x32, f32 -> f16
        for (int i = 0; i < 16; ++i) {
            int id = lane + i * 32;
            int m = id >> 5, k = id & 31;
            xs[wid][m * 32 + k] =
                (_Float16)x[(size_t)(b * SEQ + tw + m) * EMB + e0 + k];
        }
        // W chunks 32x64 staged N-major, cooperative across all 128 threads
#pragma unroll
        for (int w = 0; w < 3; ++w) {
            const float* W = wg[w];
            for (int i = 0; i < 16; ++i) {
                int id = tid + i * 128;
                int n = id & 63, k = id >> 6;
                wls[w][n * 32 + k] = (_Float16)W[(size_t)(e0 + k) * HEAD + n];
            }
        }
        __syncthreads();

        v16h a = frag_a(xs[wid], 32);
#pragma unroll
        for (int w = 0; w < 3; ++w)
#pragma unroll
            for (int nt = 0; nt < 4; ++nt) {
                v16h bf = frag_b(&wls[w][nt * 16 * 32], 32);
                acc[w][nt] = __builtin_amdgcn_wmma_f32_16x16x32_f16(
                    false, a, false, bf, (short)0, acc[w][nt], false, false);
            }
        __syncthreads();
    }

    const int nl = lane & 15;
    const int mb = (lane >> 4) * 8;
#pragma unroll
    for (int nt = 0; nt < 4; ++nt)
#pragma unroll
        for (int r = 0; r < 8; ++r) {
            size_t idx = (size_t)(b * SEQ + tw + mb + r) * HEAD + nt * 16 + nl;
            qh[idx] = (_Float16)(acc[0][nt][r] * 0.125f);  // fold 1/sqrt(64) into q
            kh[idx] = (_Float16)acc[1][nt][r];
            vh[idx] = (_Float16)acc[2][nt][r];
        }
}

// ---- Kernel 2: causal flash attention, 4 waves/block share K/V staging ------
__global__ void __launch_bounds__(128)
attn_kernel(const _Float16* __restrict__ qh,
            const _Float16* __restrict__ kh,
            const _Float16* __restrict__ vh,
            float* __restrict__ out)
{
    __shared__ alignas(16) _Float16 qs[4][16 * 64];  // per-wave Q tile [m][d]
    __shared__ alignas(16) _Float16 Kl[32 * 64];     // shared [key][d]
    __shared__ alignas(16) _Float16 Vl[64 * 32];     // shared [d][key]
    __shared__ alignas(16) _Float16 Pl[4][16 * 32];  // per-wave P staging [m][key]

    const int tid  = threadIdx.x;
    const int wid  = tid >> 5;
    const int lane = tid & 31;
    const int b    = blockIdx.x / (SEQ / 64);
    const int q0b  = (blockIdx.x % (SEQ / 64)) * 64;
    const int q0   = q0b + wid * 16;                 // this wave's query tile
    const int nl   = lane & 15;
    const int mb   = (lane >> 4) * 8;

    // async-DMA this wave's Q tile (2 KB) into LDS
    {
        uint64_t g = (uint64_t)(uintptr_t)(qh + (size_t)(b * SEQ + q0) * HEAD);
        uint32_t l = lds_lo32(&qs[wid][0]);
#pragma unroll
        for (int i = 0; i < 4; ++i) {
            uint32_t off = (uint32_t)(lane + i * 32) * 16u;
            async_b128(l + off, g + off);
        }
        wait_async0();
    }
    __syncthreads();
    v16h qa0 = frag_a(qs[wid], 64);       // d 0..31
    v16h qa1 = frag_a(qs[wid] + 32, 64);  // d 32..63

    const v8f zero = {0.f,0.f,0.f,0.f,0.f,0.f,0.f,0.f};
    v8f o[4];
#pragma unroll
    for (int t = 0; t < 4; ++t) o[t] = zero;
    float mrow[8], lrow[8];
#pragma unroll
    for (int r = 0; r < 8; ++r) { mrow[r] = -1e30f; lrow[r] = 0.f; }

    // block-uniform chunk loop; waves past their diagonal compute fully-masked
    // chunks (P == 0 exactly) so EXEC stays all-ones around every WMMA.
    for (int kb = 0; kb <= q0b + 63; kb += 32) {
        // K chunk (4 KB, row-major copy) via async DMA, all 128 threads
        {
            uint64_t g = (uint64_t)(uintptr_t)(kh + (size_t)(b * SEQ + kb) * HEAD);
            uint32_t l = lds_lo32(Kl);
#pragma unroll
            for (int i = 0; i < 2; ++i) {
                uint32_t off = (uint32_t)(tid + i * 128) * 16u;
                async_b128(l + off, g + off);
            }
        }
        // V chunk transposed -> [d][key], scalar path (needs element transpose)
        for (int i = 0; i < 16; ++i) {
            int id = tid + i * 128;
            int key = id >> 6, d = id & 63;
            Vl[d * 32 + key] = vh[(size_t)(b * SEQ + kb + key) * HEAD + d];
        }
        wait_async0();
        __syncthreads();

        // S = Q (16x64) x K^T -> two 16x16 f32 tiles
        v8f s[2];
#pragma unroll
        for (int st = 0; st < 2; ++st) {
            v16h kb0 = frag_b(Kl + (st * 16) * 64, 64);
            v16h kb1 = frag_b(Kl + (st * 16) * 64 + 32, 64);
            v8f sv = zero;
            sv = __builtin_amdgcn_wmma_f32_16x16x32_f16(false, qa0, false, kb0,
                                                        (short)0, sv, false, false);
            sv = __builtin_amdgcn_wmma_f32_16x16x32_f16(false, qa1, false, kb1,
                                                        (short)0, sv, false, false);
            s[st] = sv;
        }

        // causal mask
        if (kb + 31 > q0) {
#pragma unroll
            for (int st = 0; st < 2; ++st)
#pragma unroll
                for (int r = 0; r < 8; ++r) {
                    int kg = kb + st * 16 + nl;
                    int qg = q0 + mb + r;
                    if (kg > qg) s[st][r] = -1e30f;
                }
        }

        // online softmax; rows live across the 16 lanes of each half-wave
#pragma unroll
        for (int r = 0; r < 8; ++r) {
            float rm = fmaxf(s[0][r], s[1][r]);
            rm = fmaxf(rm, __shfl_xor(rm, 8, 32));
            rm = fmaxf(rm, __shfl_xor(rm, 4, 32));
            rm = fmaxf(rm, __shfl_xor(rm, 2, 32));
            rm = fmaxf(rm, __shfl_xor(rm, 1, 32));
            float mnew  = fmaxf(mrow[r], rm);
            float alpha = __expf(mrow[r] - mnew);
            float p0 = __expf(s[0][r] - mnew);
            float p1 = __expf(s[1][r] - mnew);
            s[0][r] = p0; s[1][r] = p1;
            float rs = p0 + p1;
            rs += __shfl_xor(rs, 8, 32);
            rs += __shfl_xor(rs, 4, 32);
            rs += __shfl_xor(rs, 2, 32);
            rs += __shfl_xor(rs, 1, 32);
            lrow[r] = lrow[r] * alpha + rs;
            mrow[r] = mnew;
#pragma unroll
            for (int t = 0; t < 4; ++t) o[t][r] *= alpha;
        }

        // P (C-layout f32) -> per-wave LDS f16 staging
#pragma unroll
        for (int st = 0; st < 2; ++st)
#pragma unroll
            for (int r = 0; r < 8; ++r)
                Pl[wid][(mb + r) * 32 + st * 16 + nl] = (_Float16)s[st][r];
        __syncthreads();

        // O += P (16x32) x V (32x64)
        v16h pa = frag_a(Pl[wid], 32);
#pragma unroll
        for (int t = 0; t < 4; ++t) {
            v16h vb = frag_b(Vl + (t * 16) * 32, 32);
            o[t] = __builtin_amdgcn_wmma_f32_16x16x32_f16(false, pa, false, vb,
                                                          (short)0, o[t], false, false);
        }
        __syncthreads();   // Kl/Vl reused next chunk
    }

    // normalize and write fp32 output
#pragma unroll
    for (int r = 0; r < 8; ++r) lrow[r] = 1.0f / lrow[r];
#pragma unroll
    for (int t = 0; t < 4; ++t)
#pragma unroll
        for (int r = 0; r < 8; ++r)
            out[(size_t)(b * SEQ + q0 + mb + r) * HEAD + t * 16 + nl] =
                o[t][r] * lrow[r];
}

extern "C" void kernel_launch(void* const* d_in, const int* in_sizes, int n_in,
                              void* d_out, int out_size, void* d_ws, size_t ws_size,
                              hipStream_t stream) {
    const float* x  = (const float*)d_in[0];
    const float* Wk = (const float*)d_in[1];
    const float* Wq = (const float*)d_in[2];
    const float* Wv = (const float*)d_in[3];

    _Float16* qh = (_Float16*)d_ws;                      // 3 x B*T*H f16 = 6 MB
    _Float16* kh = qh + (size_t)BATCH * SEQ * HEAD;
    _Float16* vh = kh + (size_t)BATCH * SEQ * HEAD;
    float* out   = (float*)d_out;

    const int nblk = BATCH * (SEQ / 64);                 // 256 blocks, 4 waves each
    qkv_proj_kernel<<<nblk, 128, 0, stream>>>(x, Wq, Wk, Wv, qh, kh, vh);
    attn_kernel<<<nblk, 128, 0, stream>>>(qh, kh, vh, out);
}
